// Du_Att_57999238365458
// MI455X (gfx1250) — matrix operand order repaired and checked by reading, not compiled, not run on hardware
//
#include <hip/hip_runtime.h>
#include <hip/hip_bf16.h>
#include <math.h>
#include <stdint.h>

// ---------------------------------------------------------------------------
// Du_Att for MI455X (gfx1250, wave32).  All GEMM-like work runs on the WMMA
// pipe via v_wmma_f32_16x16x32_f16 (f16 in, f32 accumulate).  The 16384-long
// argsort uses a single-workgroup bitonic sort in 128KB of LDS (CDNA5 WGP has
// 320KB LDS).  The TRA window tile load uses the Tensor Data Mover
// (tensor_load_to_lds + s_wait_tensorcnt) when the toolchain exposes the
// 5-arg ROCm-7.2 builtin; otherwise it falls back to plain vector loads.
// ---------------------------------------------------------------------------

#define BQ     8
#define CHALF  64
#define HH     128
#define WW     128
#define HWSZ   16384      // 128*128
#define NHEADS 8

typedef __attribute__((ext_vector_type(16))) _Float16     v16h;
typedef __attribute__((ext_vector_type(8)))  float        v8f;
typedef __attribute__((ext_vector_type(4)))  unsigned int v4u;
typedef __attribute__((ext_vector_type(8)))  int          v8i;
typedef __attribute__((ext_vector_type(4)))  int          v4i;

#if defined(__HIP_DEVICE_COMPILE__) && defined(__gfx1250__) && \
    __has_builtin(__builtin_amdgcn_tensor_load_to_lds) &&      \
    __has_builtin(__builtin_amdgcn_s_wait_tensorcnt) && (__clang_major__ < 23)
#define USE_TDM 1
#else
#define USE_TDM 0
#endif

// A-matrix (16x32 f16) half->K mapping per CDNA5 ISA:
//   lane<16 : halves 0..7 -> K0..7,  halves 8..15 -> K16..23
//   lane>=16: halves 0..7 -> K8..15, halves 8..15 -> K24..31
__device__ __forceinline__ int a_kmap(int h, int l) {
  int k = (h < 8) ? h : (h + 8);
  if (l >= 16) k += 8;
  return k;
}
// B-matrix (32x16 f16): lane gives N (= lane&15); same K order as A:
// lanes<16: K0..7,K16..23; lanes>=16: K8..15,K24..31.
__device__ __forceinline__ int b_kmap(int h, int l) {
  return a_kmap(h, l);
}

// ---------------------------------------------------------------------------
// 1) LayerNorm over 128 channels, split into halves x1 / x2 (each (B,64,HW)).
// ---------------------------------------------------------------------------
__global__ void k_ln_split(const float* __restrict__ x,
                           const float* __restrict__ lw,
                           const float* __restrict__ lb,
                           float* __restrict__ x1, float* __restrict__ x2) {
  int t = blockIdx.x * blockDim.x + threadIdx.x;     // one spatial position
  if (t >= BQ * HWSZ) return;
  int b = t >> 14, p = t & (HWSZ - 1);
  const float* px = x + (size_t)b * 128 * HWSZ + p;
  float s = 0.f, s2 = 0.f;
  for (int c = 0; c < 128; ++c) {
    float v = px[(size_t)c * HWSZ];
    s += v; s2 += v * v;
  }
  float mu  = s * (1.f / 128.f);
  float var = s2 * (1.f / 128.f) - mu * mu;
  float inv = rsqrtf(var + 1e-5f);
  for (int c = 0; c < 128; ++c) {
    float v = (px[(size_t)c * HWSZ] - mu) * inv * lw[c] + lb[c];
    if (c < CHALF) x1[((size_t)b * CHALF + c) * HWSZ + p] = v;
    else           x2[((size_t)b * CHALF + (c - CHALF)) * HWSZ + p] = v;
  }
}

// ---------------------------------------------------------------------------
// 2) Bitonic key/index sort of a 128-long column.  mode 0: along H (fixed w),
//    mode 1: along W (fixed h).  Ascending, matches jnp.sort/argsort.
// ---------------------------------------------------------------------------
__global__ void k_sort128(const float* __restrict__ src,
                          float* __restrict__ vals, int* __restrict__ idx,
                          int mode) {
  __shared__ float kv[128];
  __shared__ int   ki[128];
  int t  = threadIdx.x;
  int bc = blockIdx.x >> 7;          // b*64 + c
  int q  = blockIdx.x & 127;         // the fixed coordinate
  size_t base = (size_t)bc * HWSZ;
  size_t addr = base + (mode == 0 ? (size_t)t * WW + q : (size_t)q * WW + t);
  kv[t] = src[addr];
  ki[t] = t;
  __syncthreads();
  for (int k = 2; k <= 128; k <<= 1)
    for (int j = k >> 1; j > 0; j >>= 1) {
      int ixj = t ^ j;
      if (ixj > t) {
        bool up = ((t & k) == 0);
        float a = kv[t], c = kv[ixj];
        if ((a > c) == up) {
          kv[t] = c; kv[ixj] = a;
          int tmp = ki[t]; ki[t] = ki[ixj]; ki[ixj] = tmp;
        }
      }
      __syncthreads();
    }
  size_t oaddr = base + (mode == 0 ? (size_t)t * WW + q : (size_t)q * WW + t);
  vals[oaddr] = kv[t];
  idx[oaddr]  = ki[t];
}

// ---------------------------------------------------------------------------
// 3) Bitonic key/index sort of a 16384-long row in 128KB of dynamic LDS.
//    One workgroup (1024 threads) per (b, c) row of v = qkv2[:,128+c].
// ---------------------------------------------------------------------------
__global__ void k_sort16k(const float* __restrict__ qkv2,
                          float* __restrict__ vs, int* __restrict__ idxv) {
  extern __shared__ unsigned char sm[];
  float* kv = (float*)sm;
  int*   ki = (int*)(sm + (size_t)HWSZ * sizeof(float));
  int row = blockIdx.x;                 // b*64 + c
  int b = row >> 6, c = row & 63;
  const float* src = qkv2 + ((size_t)b * 192 + 128 + c) * HWSZ;
  for (int i = threadIdx.x; i < HWSZ; i += blockDim.x) { kv[i] = src[i]; ki[i] = i; }
  __syncthreads();
  for (int k = 2; k <= HWSZ; k <<= 1)
    for (int j = k >> 1; j > 0; j >>= 1) {
      for (int i = threadIdx.x; i < HWSZ; i += blockDim.x) {
        int ixj = i ^ j;
        if (ixj > i) {
          bool up = ((i & k) == 0);
          float a = kv[i], d = kv[ixj];
          if ((a > d) == up) {
            kv[i] = d; kv[ixj] = a;
            int tmp = ki[i]; ki[i] = ki[ixj]; ki[ixj] = tmp;
          }
        }
      }
      __syncthreads();
    }
  float* ov = vs   + (size_t)row * HWSZ;
  int*   oi = idxv + (size_t)row * HWSZ;
  for (int i = threadIdx.x; i < HWSZ; i += blockDim.x) { ov[i] = kv[i]; oi[i] = ki[i]; }
}

// ---------------------------------------------------------------------------
// 4) Gather q,k with the v argsort permutation.
// ---------------------------------------------------------------------------
__global__ void k_gather2(const float* __restrict__ qkv2,
                          const int* __restrict__ idxv,
                          float* __restrict__ qg, float* __restrict__ kg) {
  size_t t = (size_t)blockIdx.x * blockDim.x + threadIdx.x;
  if (t >= (size_t)BQ * CHALF * HWSZ) return;
  size_t row = t >> 14;
  size_t j   = t & (HWSZ - 1);
  int b = (int)(row >> 6), c = (int)(row & 63);
  int src = idxv[row * HWSZ + j];
  qg[row * HWSZ + j] = qkv2[((size_t)b * 192 +      c) * HWSZ + src];
  kg[row * HWSZ + j] = qkv2[((size_t)b * 192 + 64 + c) * HWSZ + src];
}

// ---------------------------------------------------------------------------
// 5) WMMA GEMM: Y[b, yoff+m, p] = sum_k W[m,k] * X[b,k,p] + bias[m] (+resid).
//    Each wave owns one 16x16 tile; K=64 -> two v_wmma_f32_16x16x32_f16.
// ---------------------------------------------------------------------------
__global__ void k_gemm_wmma(const float* __restrict__ Wm,
                            const float* __restrict__ bias,
                            const float* __restrict__ X,
                            float* __restrict__ Y,
                            const float* __restrict__ resid,
                            int M, int K, int NT, int ycstride, int ycoff) {
  int wave = (blockIdx.x * blockDim.x + threadIdx.x) >> 5;
  int l    = threadIdx.x & 31;
  int mt = wave / NT;
  int nt = wave - mt * NT;
  int n0 = nt * 16;
  int b  = n0 >> 14;
  int pp = (n0 & (HWSZ - 1)) + (l & 15);
  int m  = mt * 16 + (l & 15);

  v16h a0, a1, b0, b1;
  const float* wrow = Wm + (size_t)m * K;
  for (int h = 0; h < 16; ++h) {
    int k = a_kmap(h, l);
    a0[h] = (_Float16)wrow[k];
    a1[h] = (_Float16)wrow[k + 32];
  }
  const float* xb = X + (size_t)b * K * HWSZ + pp;
  for (int h = 0; h < 16; ++h) {
    int k = b_kmap(h, l);
    b0[h] = (_Float16)xb[(size_t)k * HWSZ];
    b1[h] = (_Float16)xb[(size_t)(k + 32) * HWSZ];
  }
  v8f c = {};
  c = __builtin_amdgcn_wmma_f32_16x16x32_f16(false, a0, false, b0, (short)0, c, false, false);
  c = __builtin_amdgcn_wmma_f32_16x16x32_f16(false, a1, false, b1, (short)0, c, false, false);

  for (int r = 0; r < 8; ++r) {
    int mm = mt * 16 + r + 8 * (l >> 4);
    float v = c[r] + bias[mm];
    size_t o = ((size_t)b * ycstride + ycoff + mm) * HWSZ + pp;
    if (resid) v += resid[o];
    Y[o] = v;
  }
}

// ---------------------------------------------------------------------------
// 6) Depthwise 3x3 conv, SAME padding, 192 channels.
// ---------------------------------------------------------------------------
__global__ void k_dw3x3(const float* __restrict__ in,
                        const float* __restrict__ w,
                        const float* __restrict__ bias,
                        float* __restrict__ out) {
  size_t t = (size_t)blockIdx.x * blockDim.x + threadIdx.x;
  if (t >= (size_t)BQ * 192 * HWSZ) return;
  int p = (int)(t & (HWSZ - 1));
  size_t bc = t >> 14;
  int ch = (int)(bc % 192);
  int y = p >> 7, xx = p & 127;
  const float* base = in + bc * HWSZ;
  const float* wc   = w + (size_t)ch * 9;
  float acc = bias[ch];
  for (int dy = -1; dy <= 1; ++dy) {
    int yy = y + dy;
    if (yy < 0 || yy >= HH) continue;
    for (int dx = -1; dx <= 1; ++dx) {
      int xc = xx + dx;
      if (xc < 0 || xc >= WW) continue;
      acc += base[(size_t)yy * WW + xc] * wc[(dy + 1) * 3 + (dx + 1)];
    }
  }
  out[t] = acc;
}

// ---------------------------------------------------------------------------
// 7) CCA attention: one block (256 thr / 8 waves) per (b, head).
//    q,k,v rows are 64 x 2048; attn = l2n(q) l2n(k)^T * temp -> softmax -> *v.
// ---------------------------------------------------------------------------
__global__ void k_cca_attn(const float* __restrict__ qg,
                           const float* __restrict__ kg,
                           const float* __restrict__ vs,
                           const float* __restrict__ temp,
                           float* __restrict__ o) {
  __shared__ float attnS[64 * 64];
  __shared__ float qinv[64], kinv[64];
  __shared__ float red[256];
  int b  = blockIdx.x >> 3;
  int hd = blockIdx.x & 7;
  int t  = threadIdx.x;
  int wv = t >> 5, l = t & 31;

  // row base for logical row i in [0,64): channel hd*8 + i/8, offset (i%8)*2048
  int r = t >> 2, part = t & 3;
  size_t qrow = ((size_t)b * CHALF + hd * 8 + (r >> 3)) * HWSZ + (size_t)(r & 7) * 2048;
  float s = 0.f;
  for (int f = part * 512; f < (part + 1) * 512; ++f) { float v = qg[qrow + f]; s += v * v; }
  red[t] = s; __syncthreads();
  if (part == 0) qinv[r] = 1.f / fmaxf(sqrtf(red[t] + red[t + 1] + red[t + 2] + red[t + 3]), 1e-12f);
  __syncthreads();
  s = 0.f;
  for (int f = part * 512; f < (part + 1) * 512; ++f) { float v = kg[qrow + f]; s += v * v; }
  red[t] = s; __syncthreads();
  if (part == 0) kinv[r] = 1.f / fmaxf(sqrtf(red[t] + red[t + 1] + red[t + 2] + red[t + 3]), 1e-12f);
  __syncthreads();

  float tp = temp[hd];
  // attn = qn * kn^T : 4x4 tiles of 16x16, K = 2048 (64 WMMA steps)
  for (int tile = wv; tile < 16; tile += 8) {
    int mt = tile >> 2, nt = tile & 3;
    int m  = mt * 16 + (l & 15);
    int nn = nt * 16 + (l & 15);
    size_t arow = ((size_t)b * CHALF + hd * 8 + (m  >> 3)) * HWSZ + (size_t)(m  & 7) * 2048;
    size_t brow = ((size_t)b * CHALF + hd * 8 + (nn >> 3)) * HWSZ + (size_t)(nn & 7) * 2048;
    float ai = qinv[m], bi = kinv[nn];
    v8f c = {};
    for (int ks = 0; ks < 2048; ks += 32) {
      if (ks + 64 < 2048) {
        __builtin_prefetch(qg + arow + ks + 64, 0, 1);
        __builtin_prefetch(kg + brow + ks + 64, 0, 1);
      }
      v16h af, bf;
      for (int h = 0; h < 16; ++h) af[h] = (_Float16)(qg[arow + ks + a_kmap(h, l)] * ai);
      for (int h = 0; h < 16; ++h) bf[h] = (_Float16)(kg[brow + ks + b_kmap(h, l)] * bi);
      c = __builtin_amdgcn_wmma_f32_16x16x32_f16(false, af, false, bf, (short)0, c, false, false);
    }
    for (int rr = 0; rr < 8; ++rr) {
      int mm = mt * 16 + rr + 8 * (l >> 4);
      int nc = nt * 16 + (l & 15);
      attnS[mm * 64 + nc] = c[rr] * tp;
    }
  }
  __syncthreads();

  if (t < 64) {                          // softmax per row
    float mx = -1e30f;
    for (int j = 0; j < 64; ++j) mx = fmaxf(mx, attnS[t * 64 + j]);
    float sum = 0.f;
    for (int j = 0; j < 64; ++j) { float e = __expf(attnS[t * 64 + j] - mx); attnS[t * 64 + j] = e; sum += e; }
    float is = 1.f / sum;
    for (int j = 0; j < 64; ++j) attnS[t * 64 + j] *= is;
  }
  __syncthreads();

  // out = attn(64x64) * v(64x2048):  wave wv -> mt = wv/2, nt = (wv&1)+2*i
  int mt = wv >> 1;
  int m2 = mt * 16 + (l & 15);
  v16h A0, A1;
  for (int h = 0; h < 16; ++h) {
    int k = a_kmap(h, l);
    A0[h] = (_Float16)attnS[m2 * 64 + k];
    A1[h] = (_Float16)attnS[m2 * 64 + k + 32];
  }
  for (int nt = (wv & 1); nt < 128; nt += 2) {
    int f0 = nt * 16 + (l & 15);
    v16h B0, B1;
    for (int h = 0; h < 16; ++h) {
      int k  = b_kmap(h, l);
      int k2 = k + 32;
      size_t v1 = ((size_t)b * CHALF + hd * 8 + (k  >> 3)) * HWSZ + (size_t)(k  & 7) * 2048;
      size_t v2 = ((size_t)b * CHALF + hd * 8 + (k2 >> 3)) * HWSZ + (size_t)(k2 & 7) * 2048;
      B0[h] = (_Float16)vs[v1 + f0];
      B1[h] = (_Float16)vs[v2 + f0];
    }
    v8f c = {};
    c = __builtin_amdgcn_wmma_f32_16x16x32_f16(false, A0, false, B0, (short)0, c, false, false);
    c = __builtin_amdgcn_wmma_f32_16x16x32_f16(false, A1, false, B1, (short)0, c, false, false);
    for (int rr = 0; rr < 8; ++rr) {
      int mm = mt * 16 + rr + 8 * (l >> 4);
      int ff = nt * 16 + (l & 15);
      size_t orow = ((size_t)b * CHALF + hd * 8 + (mm >> 3)) * HWSZ + (size_t)(mm & 7) * 2048;
      o[orow + ff] = c[rr];
    }
  }
}

// ---------------------------------------------------------------------------
// 8) Inverse permutations (scatter form of take_along_axis(.., argsort(idx))).
// ---------------------------------------------------------------------------
__global__ void k_scatter_hw(const float* __restrict__ o,
                             const int* __restrict__ idxv,
                             float* __restrict__ dst) {
  size_t t = (size_t)blockIdx.x * blockDim.x + threadIdx.x;
  if (t >= (size_t)BQ * CHALF * HWSZ) return;
  size_t row = t >> 14;
  size_t j   = t & (HWSZ - 1);
  dst[row * HWSZ + idxv[row * HWSZ + j]] = o[row * HWSZ + j];
}

__global__ void k_scatter_w(const float* __restrict__ src,
                            const int* __restrict__ idxw,
                            float* __restrict__ dst) {
  size_t t = (size_t)blockIdx.x * blockDim.x + threadIdx.x;
  if (t >= (size_t)BQ * CHALF * HWSZ) return;
  size_t rowbase = (t >> 7) << 7;         // (b*64*128 + bc*128 + h) * 128
  int j = (int)(t & 127);
  dst[rowbase + idxw[rowbase + j]] = src[rowbase + j];
}

__global__ void k_scatter_h_resid(const float* __restrict__ src,
                                  const int* __restrict__ idxh,
                                  const float* __restrict__ x,
                                  float* __restrict__ out) {
  size_t t = (size_t)blockIdx.x * blockDim.x + threadIdx.x;
  if (t >= (size_t)BQ * CHALF * HWSZ) return;
  size_t bc = t >> 14;
  int p = (int)(t & (HWSZ - 1));
  int j = p >> 7, wcol = p & 127;
  int hp = idxh[bc * HWSZ + (size_t)j * WW + wcol];
  int b = (int)(bc >> 6), c = (int)(bc & 63);
  size_t od = ((size_t)b * 128 + c) * HWSZ + (size_t)hp * WW + wcol;
  out[od] = src[bc * HWSZ + p] + x[od];
}

// ---------------------------------------------------------------------------
// 9) TRA window attention: one wave per (window, head).  seq=64, dim=8
//    (padded to K=32 for WMMA).  The q/k/v window tile (24 channels x 8x8
//    pixels) is staged into LDS by the Tensor Data Mover as a 3D tile:
//    X=8 contiguous, Y=8 rows (stride 128), Z=24 channels (stride 16384).
// ---------------------------------------------------------------------------
#define QV(pos, c3) tile[(c3) * 64 + (pos)]
#define KV(pos, c3) tile[((c3) + 8) * 64 + (pos)]
#define VV(pos, c3) tile[((c3) + 16) * 64 + (pos)]

__global__ void k_tra_attn(const float* __restrict__ qkvt,
                           const float* __restrict__ rpb,
                           const float* __restrict__ lscale,
                           float* __restrict__ otra) {
  __shared__ float tile[24 * 64];        // [c3][pos], pos = yy*8+xx
  __shared__ float attnS[64 * 64];
  __shared__ float qi[64], kiv[64];
  int win = blockIdx.x >> 3;
  int hd  = blockIdx.x & 7;
  int b = win >> 8, ih = (win >> 4) & 15, iw = win & 15;
  int l = threadIdx.x;

  const float* gbase =
      qkvt + ((size_t)b * 192 + hd * 24) * HWSZ + (size_t)(ih * 8) * WW + iw * 8;
#if USE_TDM
  {
    // Tensor DMA descriptor (CDNA5 ISA), data_size=4B.
    uint64_t ga  = (uint64_t)(uintptr_t)gbase;
    uint32_t lds = (uint32_t)(uintptr_t)(&tile[0]);   // low 32 bits of the LDS address
    v4u g0 = { 1u,                                    // count=1, load, no gather
               lds,                                   // lds_addr (bytes)
               (uint32_t)ga,                          // global_addr[31:0]
               (uint32_t)((ga >> 32) & 0x01FFFFFFu) | (2u << 30) }; // +type=2
    v8i g1 = { (int)0x00020000,                       // data_size=2 (4 bytes)
               (int)(128u << 16),                     // tensor_dim0 = 128
               (int)(128u << 16),                     // tensor_dim1 = 128
               (int)(8u << 16),                       // tile_dim0   = 8
               (int)((24u << 16) | 8u),               // tile_dim2=24, tile_dim1=8
               128,                                   // tensor_dim0_stride = 128
               (int)(0x4000u << 16),                  // tensor_dim1_stride = 16384
               0 };
    v4i g2 = { 24, 0, 0, 0 };                         // tensor_dim2 = 24
    v4i g3 = { 0, 0, 0, 0 };
    __builtin_amdgcn_tensor_load_to_lds(g0, g1, g2, g3, 0);
    __builtin_amdgcn_s_wait_tensorcnt(0);
  }
#else
  for (int e = l; e < 24 * 64; e += 32) {
    int c3 = e >> 6, pos = e & 63;
    int yy = pos >> 3, xx = pos & 7;
    tile[e] = gbase[(size_t)c3 * HWSZ + (size_t)yy * WW + xx];
  }
#endif
  __syncthreads();

  for (int rr = l; rr < 64; rr += 32) {
    float s = 0.f; for (int c = 0; c < 8; ++c) s += QV(rr, c) * QV(rr, c);
    qi[rr] = 1.f / fmaxf(sqrtf(s), 1e-12f);
    s = 0.f;      for (int c = 0; c < 8; ++c) s += KV(rr, c) * KV(rr, c);
    kiv[rr] = 1.f / fmaxf(sqrtf(s), 1e-12f);
  }
  __syncthreads();

  float ls = __expf(fminf(lscale[hd], 4.6051701860f));   // log(1/0.01)
  for (int tl = 0; tl < 16; ++tl) {
    int mt = tl >> 2, nt = tl & 3;
    int m = mt * 16 + (l & 15);
    int n = nt * 16 + (l & 15);
    v16h af, bf;
    for (int h = 0; h < 16; ++h) {
      int k = a_kmap(h, l);
      af[h] = (_Float16)((k < 8) ? QV(m, k) * qi[m] : 0.f);
    }
    for (int h = 0; h < 16; ++h) {
      int k = b_kmap(h, l);
      bf[h] = (_Float16)((k < 8) ? KV(n, k) * kiv[n] : 0.f);
    }
    v8f c = {};
    c = __builtin_amdgcn_wmma_f32_16x16x32_f16(false, af, false, bf, (short)0, c, false, false);
    for (int rr = 0; rr < 8; ++rr) {
      int mm = mt * 16 + rr + 8 * (l >> 4);
      int nc = nt * 16 + (l & 15);
      int yi = mm >> 3, xi = mm & 7, yj = nc >> 3, xj = nc & 7;
      int ridx = (yi - yj + 7) * 15 + (xi - xj + 7);
      attnS[mm * 64 + nc] = c[rr] * ls + rpb[ridx * NHEADS + hd];
    }
  }
  __syncthreads();
  for (int rr = l; rr < 64; rr += 32) {
    float mx = -1e30f;
    for (int j = 0; j < 64; ++j) mx = fmaxf(mx, attnS[rr * 64 + j]);
    float sum = 0.f;
    for (int j = 0; j < 64; ++j) { float e = __expf(attnS[rr * 64 + j] - mx); attnS[rr * 64 + j] = e; sum += e; }
    float is = 1.f / sum;
    for (int j = 0; j < 64; ++j) attnS[rr * 64 + j] *= is;
  }
  __syncthreads();

  // out(64x8) = attn(64x64) * v(64x8), N padded to 16, K=64 -> 2 WMMA steps
  v16h B0, B1;
  int n = l & 15;
  for (int h = 0; h < 16; ++h) {
    int k = b_kmap(h, l);
    B0[h] = (_Float16)((n < 8) ? VV(k, n)      : 0.f);
    B1[h] = (_Float16)((n < 8) ? VV(k + 32, n) : 0.f);
  }
  for (int mt = 0; mt < 4; ++mt) {
    int m = mt * 16 + (l & 15);
    v16h A0, A1;
    for (int h = 0; h < 16; ++h) {
      int k = a_kmap(h, l);
      A0[h] = (_Float16)attnS[m * 64 + k];
      A1[h] = (_Float16)attnS[m * 64 + k + 32];
    }
    v8f c = {};
    c = __builtin_amdgcn_wmma_f32_16x16x32_f16(false, A0, false, B0, (short)0, c, false, false);
    c = __builtin_amdgcn_wmma_f32_16x16x32_f16(false, A1, false, B1, (short)0, c, false, false);
    for (int rr = 0; rr < 8; ++rr) {
      int mm = mt * 16 + rr + 8 * (l >> 4);
      int nn = l & 15;
      if (nn < 8) {
        int yy = mm >> 3, xx = mm & 7;
        otra[((size_t)b * CHALF + hd * 8 + nn) * HWSZ +
             (size_t)(ih * 8 + yy) * WW + iw * 8 + xx] = c[rr];
      }
    }
  }
}

// ---------------------------------------------------------------------------
// Launch sequence (all on `stream`, graph-capture safe).
// ---------------------------------------------------------------------------
extern "C" void kernel_launch(void* const* d_in, const int* in_sizes, int n_in,
                              void* d_out, int out_size, void* d_ws, size_t ws_size,
                              hipStream_t stream) {
  const float* x        = (const float*)d_in[0];
  const float* ln_w     = (const float*)d_in[1];
  const float* ln_b     = (const float*)d_in[2];
  const float* cca_temp = (const float*)d_in[3];
  const float* cca_qw   = (const float*)d_in[4];
  const float* cca_qb   = (const float*)d_in[5];
  const float* cca_dww  = (const float*)d_in[6];
  const float* cca_dwb  = (const float*)d_in[7];
  const float* tra_qw   = (const float*)d_in[8];
  const float* tra_qb   = (const float*)d_in[9];
  const float* tra_rpb  = (const float*)d_in[10];
  const float* tra_ls   = (const float*)d_in[11];
  const float* tra_pw   = (const float*)d_in[12];
  const float* tra_pb   = (const float*)d_in[13];
  float* out = (float*)d_out;

  const size_t S = (size_t)BQ * CHALF * HWSZ;   // 8,388,608 floats
  float* ws   = (float*)d_ws;
  float* x1   = ws + 0 * S;
  float* x2   = ws + 1 * S;
  float* xsh  = ws + 2 * S;
  int*   idxh = (int*)(ws + 3 * S);
  float* xs   = ws + 4 * S;
  int*   idxw = (int*)(ws + 5 * S);
  float* qg   = ws + 6 * S;
  float* kg   = ws + 7 * S;
  float* qkv1 = ws + 8 * S;     // 3S
  float* qkv2 = ws + 11 * S;    // 3S
  // dead-buffer aliases (stream is sequential, lifetimes verified):
  float* vsb  = x1;             // x1 dead after sort-H
  int*   idxv = (int*)xsh;      // xsh dead after sort-W
  float* tmp1 = xs;             // xs dead after CCA qkv GEMM
  float* tmp2 = qg;             // qg dead after CCA attention
  float* qkvt = qkv1;           // free again after CCA scatters
  float* otra = x1;             // vs dead after CCA attention

  const int NT_BHW = (BQ * HWSZ) / 16;          // 8192
  (void)in_sizes; (void)n_in; (void)out_size; (void)ws_size;

  // allow 128KB dynamic LDS for the big sort (host-side, capture-safe)
  hipFuncSetAttribute(reinterpret_cast<const void*>(k_sort16k),
                      hipFuncAttributeMaxDynamicSharedMemorySize, 131072);

  // 1) LayerNorm + split
  k_ln_split<<<(BQ * HWSZ) / 256, 256, 0, stream>>>(x, ln_w, ln_b, x1, x2);

  // -------- CCA half --------
  // 2) sort along H, then along W
  k_sort128<<<BQ * CHALF * WW, 128, 0, stream>>>(x1, xsh, idxh, 0);
  k_sort128<<<BQ * CHALF * HH, 128, 0, stream>>>(xsh, xs, idxw, 1);
  // 3) qkv 1x1 GEMM (WMMA): (192x64) @ (64 x B*HW)
  k_gemm_wmma<<<(12 * NT_BHW) / 8, 256, 0, stream>>>(cca_qw, cca_qb, xs, qkv1,
                                                     nullptr, 192, 64, NT_BHW, 192, 0);
  // 4) depthwise 3x3
  k_dw3x3<<<(BQ * 192 * HWSZ) / 256, 256, 0, stream>>>(qkv1, cca_dww, cca_dwb, qkv2);
  // 5) argsort v over 16384 (128KB LDS bitonic), then gather q,k
  k_sort16k<<<BQ * CHALF, 1024, 131072, stream>>>(qkv2, vsb, idxv);
  k_gather2<<<(int)(S / 256), 256, 0, stream>>>(qkv2, idxv, qg, kg);
  // 6) attention per (b, head)
  k_cca_attn<<<BQ * NHEADS, 256, 0, stream>>>(qg, kg, vsb, cca_temp, qkv1);
  // 7) inverse permutations + residual into out channels [0,64)
  k_scatter_hw<<<(int)(S / 256), 256, 0, stream>>>(qkv1, idxv, tmp1);
  k_scatter_w <<<(int)(S / 256), 256, 0, stream>>>(tmp1, idxw, tmp2);
  k_scatter_h_resid<<<(int)(S / 256), 256, 0, stream>>>(tmp2, idxh, x, out);

  // -------- TRA half --------
  // 8) qkv GEMM (WMMA)
  k_gemm_wmma<<<(12 * NT_BHW) / 8, 256, 0, stream>>>(tra_qw, tra_qb, x2, qkvt,
                                                     nullptr, 192, 64, NT_BHW, 192, 0);
  // 9) window attention: one wave per (window, head), TDM tile staging
  k_tra_attn<<<BQ * 16 * 16 * NHEADS, 32, 0, stream>>>(qkvt, tra_rpb, tra_ls, otra);
  // 10) output projection (WMMA) + residual into out channels [64,128)
  k_gemm_wmma<<<(4 * NT_BHW) / 8, 256, 0, stream>>>(tra_pw, tra_pb, otra, out,
                                                    x, 64, 64, NT_BHW, 128, 64);
}
